// SpatialConv_188978561174
// MI455X (gfx1250) — matrix operand order repaired and checked by reading, not compiled, not run on hardware
//
#include <hip/hip_runtime.h>
#include <hip/hip_bf16.h>

typedef __attribute__((ext_vector_type(16))) __bf16 v16bf;
typedef __attribute__((ext_vector_type(8)))  float  v8f;

#define HID 128
#define EMB 32
#define EIN 32

static __device__ __forceinline__ __bf16 f2bf(float f) {
  unsigned u = __builtin_bit_cast(unsigned, f);
  unsigned r = u + 0x7FFFu + ((u >> 16) & 1u);   // round-to-nearest-even
  unsigned short h = (unsigned short)(r >> 16);
  return __builtin_bit_cast(__bf16, h);
}

// monotone float -> uint key for atomicMax on floats (key 0 == "-NaN" == minimum)
static __device__ __forceinline__ unsigned fkey(float f) {
  unsigned u = __builtin_bit_cast(unsigned, f);
  return (u & 0x80000000u) ? ~u : (u | 0x80000000u);
}
static __device__ __forceinline__ float funkey(unsigned k) {
  unsigned u = (k & 0x80000000u) ? (k ^ 0x80000000u) : ~k;
  return __builtin_bit_cast(float, u);
}

// ---------------------------------------------------------------------------
// Pack a row-major [K x 128] f32 weight into CDNA5 WMMA B-fragments (bf16).
// Fragment f = kt*8 + nt is a 32x16 (KxN) tile; per lane: N = lane%16,
// klocal = (lane<16 ? i : i+16), i = 0..15 packed contiguously (32B per lane).
// ---------------------------------------------------------------------------
__global__ void pack_b_kernel(const float* __restrict__ W, __bf16* __restrict__ out, int K) {
  int idx = blockIdx.x * blockDim.x + threadIdx.x;
  int total = (K / 32) * 8 * 512;
  if (idx >= total) return;
  int f    = idx >> 9;
  int r    = idx & 511;
  int lane = r >> 4;
  int i    = r & 15;
  int kt = f >> 3, nt = f & 7;
  int nloc   = lane & 15;
  int klocal = (lane < 16) ? i : (i + 16);
  int kg     = kt * 32 + klocal;
  out[idx] = f2bf(W[kg * 128 + nt * 16 + nloc]);
}

// ---------------------------------------------------------------------------
// Fused edge-MLP + Q/K/V projection. 8 waves/block, one 16-edge tile per wave.
// ---------------------------------------------------------------------------
__global__ __launch_bounds__(256)
void fused_mlp_qkv(const float* __restrict__ x, const int* __restrict__ ei,
                   const float* __restrict__ ea,
                   const float* __restrict__ W1, const float* __restrict__ b1,
                   const __bf16* __restrict__ pW2, const float* __restrict__ b2,
                   const __bf16* __restrict__ pWq, const __bf16* __restrict__ pWk,
                   const __bf16* __restrict__ pWv,
                   float* __restrict__ q, float* __restrict__ k, float* __restrict__ v,
                   int E) {
  __shared__ float  sW1[34 * 32];
  __shared__ float  sb1[32];
  __shared__ float  sb2[128];
  __shared__ __bf16 hA[8][16 * 32];    // first-layer activations (per wave)
  __shared__ __bf16 hB[8][16 * 128];   // post-MLP h tile (per wave)

  const int tid  = threadIdx.x;
  const int wave = tid >> 5;
  const int lane = tid & 31;

  for (int i = tid; i < 34 * 32; i += 256) sW1[i] = W1[i];
  for (int i = tid; i < 32;      i += 256) sb1[i] = b1[i];
  for (int i = tid; i < 128;     i += 256) sb2[i] = b2[i];
  __syncthreads();

  const int e0 = (blockIdx.x * 8 + wave) * 16;
  const int m  = lane & 15;
  const int kb = (lane < 16) ? 0 : 16;

  // ---- layer 1: h1 = relu([xs,xd,ea] @ W1 + b1), 2 lanes per edge ----
  int ec = e0 + m; if (ec >= E) ec = E - 1;
  const int   si = ei[ec], di = ei[E + ec];
  const float xs = x[si],  xd = x[di];
  float acc[16];
#pragma unroll
  for (int j = 0; j < 16; j++)
    acc[j] = sb1[kb + j] + xs * sW1[kb + j] + xd * sW1[32 + kb + j];
  for (int jn = 0; jn < EIN; jn++) {
    float a = ea[(long)ec * EIN + jn];
#pragma unroll
    for (int j = 0; j < 16; j++) acc[j] += a * sW1[(2 + jn) * 32 + kb + j];
  }
#pragma unroll
  for (int j = 0; j < 16; j++) {
    float r = acc[j] > 0.f ? acc[j] : 0.f;
    hA[wave][m * 32 + kb + j] = f2bf(r);
  }
  __syncthreads();

  // ---- A fragment (16x32 bf16, CDNA5 layout) ----
  v16bf afrag;
#pragma unroll
  for (int i = 0; i < 16; i++) {
    int kl = kb + i;
    afrag[i] = hA[wave][m * 32 + kl];
  }

  // ---- layer 2: h = h1 @ W2 + b2 via 8 WMMAs, write bf16 tile to LDS ----
  const int n      = lane & 15;
  const int rowoff = (lane < 16) ? 0 : 8;
#pragma unroll
  for (int nt = 0; nt < 8; nt++) {
    float bias = sb2[nt * 16 + n];
    v8f c;
#pragma unroll
    for (int r = 0; r < 8; r++) c[r] = bias;
    v16bf bfrag = *(const v16bf*)(pW2 + nt * 512 + lane * 16);
    c = __builtin_amdgcn_wmma_f32_16x16x32_bf16(false, afrag, false, bfrag,
                                                (short)0, c, false, false);
#pragma unroll
    for (int r = 0; r < 8; r++)
      hB[wave][(rowoff + r) * 128 + nt * 16 + n] = f2bf(c[r]);
  }
  __syncthreads();

  // ---- re-fragment h as 4 K-tiles of 16x32 ----
  v16bf A2[4];
#pragma unroll
  for (int kt = 0; kt < 4; kt++) {
#pragma unroll
    for (int i = 0; i < 16; i++) {
      int kl = kb + i;
      A2[kt][i] = hB[wave][m * 128 + kt * 32 + kl];
    }
  }

  // ---- q/k/v = h @ W, 32 WMMAs each ----
  const __bf16* packs[3] = {pWq, pWk, pWv};
  float*        outs[3]  = {q, k, v};
#pragma unroll
  for (int p = 0; p < 3; p++) {
    const __bf16* pW  = packs[p];
    float*        out = outs[p];
#pragma unroll
    for (int nt = 0; nt < 8; nt++) {
      v8f accq = {};
#pragma unroll
      for (int kt = 0; kt < 4; kt++) {
        v16bf bfrag = *(const v16bf*)(pW + (kt * 8 + nt) * 512 + lane * 16);
        accq = __builtin_amdgcn_wmma_f32_16x16x32_bf16(false, A2[kt], false, bfrag,
                                                       (short)0, accq, false, false);
      }
#pragma unroll
      for (int r = 0; r < 8; r++) {
        int e = e0 + rowoff + r;
        if (e < E) out[(long)e * HID + nt * 16 + n] = accq[r];
      }
    }
  }
}

// ---------------------------------------------------------------------------
// Edge-to-edge attention: wave32 per e2e edge.
// ---------------------------------------------------------------------------
__global__ void attn_logits(const float* __restrict__ q, const float* __restrict__ k,
                            const int* __restrict__ e2e, float* __restrict__ logits,
                            unsigned* __restrict__ segmax, int EE) {
  int wave = threadIdx.x >> 5, lane = threadIdx.x & 31;
  long ee = (long)blockIdx.x * 8 + wave;
  if (ee >= EE) return;
  int es = e2e[ee], ed = e2e[EE + ee];
  float4 qv = ((const float4*)(q + (long)ed * HID))[lane];
  float4 kv = ((const float4*)(k + (long)es * HID))[lane];
  float p = qv.x * kv.x + qv.y * kv.y + qv.z * kv.z + qv.w * kv.w;
  p += __shfl_xor(p, 1, 32);
  p += __shfl_xor(p, 2, 32);
  p += __shfl_xor(p, 4, 32);
  if ((lane & 7) == 0) {
    int h = lane >> 3;
    float lg = p * 0.17677669529663687f;   // 1/sqrt(32)
    logits[ee * 4 + h] = lg;
    atomicMax(&segmax[(long)ed * 4 + h], fkey(lg));
  }
}

__global__ void attn_exp(const int* __restrict__ e2e, float* __restrict__ logits,
                         const unsigned* __restrict__ segmax, float* __restrict__ segsum,
                         int EE) {
  long i = (long)blockIdx.x * blockDim.x + threadIdx.x;
  if (i >= (long)EE * 4) return;
  long ee = i >> 2; int h = (int)(i & 3);
  int ed = e2e[EE + ee];
  float m = funkey(segmax[(long)ed * 4 + h]);
  float e = __expf(logits[i] - m);
  logits[i] = e;
  atomicAdd(&segsum[(long)ed * 4 + h], e);
}

__global__ void attn_agg(const float* __restrict__ v, const int* __restrict__ e2e,
                         const float* __restrict__ logits, const float* __restrict__ segsum,
                         float* __restrict__ agg, int EE) {
  int wave = threadIdx.x >> 5, lane = threadIdx.x & 31;
  long ee = (long)blockIdx.x * 8 + wave;
  if (ee >= EE) return;
  int es = e2e[ee], ed = e2e[EE + ee];
  int h = lane >> 3;
  float e = logits[ee * 4 + h];
  float s = segsum[(long)ed * 4 + h];
  float alpha = e / (s + 1e-16f);
  float4 vv = ((const float4*)(v + (long)es * HID))[lane];
  float* dst = agg + (long)ed * HID + lane * 4;
  atomicAdd(dst + 0, alpha * vv.x);
  atomicAdd(dst + 1, alpha * vv.y);
  atomicAdd(dst + 2, alpha * vv.z);
  atomicAdd(dst + 3, alpha * vv.w);
}

__global__ void uef_mean(const float* __restrict__ agg, float* __restrict__ out_uef, int E) {
  int e = blockIdx.x * blockDim.x + threadIdx.x;
  if (e >= E) return;
  const float4* row = (const float4*)(agg + (long)e * HID);
  float acc = 0.f;
#pragma unroll
  for (int i = 0; i < 32; i++) {
    float4 t = row[i];
    acc += t.x + t.y + t.z + t.w;
  }
  out_uef[e] = acc * (1.0f / 128.0f);
}

// ---------------------------------------------------------------------------
// Node-to-node part (algebraically reduced to scalars).
// coef[0..3] = sum_d Wq2[h,d]*Wk2[h,d];  coef[4..7] = sum_d Wv2[h,d]
// ---------------------------------------------------------------------------
__global__ void coef_kernel(const float* __restrict__ Wq2, const float* __restrict__ Wk2,
                            const float* __restrict__ Wv2, float* __restrict__ coef) {
  int h = threadIdx.x;
  if (h < 4) {
    float dqk = 0.f, sv = 0.f;
    for (int d = 0; d < 32; d++) {
      dqk += Wq2[h * 32 + d] * Wk2[h * 32 + d];
      sv  += Wv2[h * 32 + d];
    }
    coef[h] = dqk;
    coef[4 + h] = sv;
  }
}

__global__ void n2n_max(const float* __restrict__ x, const int* __restrict__ n2n,
                        const float* __restrict__ coef, unsigned* __restrict__ nmax, int C) {
  long c = (long)blockIdx.x * blockDim.x + threadIdx.x;
  if (c >= C) return;
  int ns = n2n[c], nd = n2n[C + c];
  float p = x[ns] * x[nd] * 0.17677669529663687f;
#pragma unroll
  for (int h = 0; h < 4; h++)
    atomicMax(&nmax[(long)nd * 4 + h], fkey(p * coef[h]));
}

__global__ void n2n_sum(const float* __restrict__ x, const int* __restrict__ n2n,
                        const float* __restrict__ coef, const unsigned* __restrict__ nmax,
                        float* __restrict__ nsum, int C) {
  long c = (long)blockIdx.x * blockDim.x + threadIdx.x;
  if (c >= C) return;
  int ns = n2n[c], nd = n2n[C + c];
  float p = x[ns] * x[nd] * 0.17677669529663687f;
#pragma unroll
  for (int h = 0; h < 4; h++) {
    float m = funkey(nmax[(long)nd * 4 + h]);
    atomicAdd(&nsum[(long)nd * 4 + h], __expf(p * coef[h] - m));
  }
}

__global__ void n2n_msg(const float* __restrict__ x, const int* __restrict__ n2n,
                        const float* __restrict__ coef, const unsigned* __restrict__ nmax,
                        const float* __restrict__ nsum, float* __restrict__ out, int C) {
  long c = (long)blockIdx.x * blockDim.x + threadIdx.x;
  if (c >= C) return;
  int ns = n2n[c], nd = n2n[C + c];
  float xns = x[ns], xnd = x[nd];
  float p = xns * xnd * 0.17677669529663687f;
  float msg = 0.f;
#pragma unroll
  for (int h = 0; h < 4; h++) {
    float m = funkey(nmax[(long)nd * 4 + h]);
    float s = nsum[(long)nd * 4 + h];
    float e = __expf(p * coef[h] - m);
    msg += (e / (s + 1e-16f)) * coef[4 + h];
  }
  msg *= xns * (1.0f / 128.0f);
  atomicAdd(&out[nd], msg);
}

// ---------------------------------------------------------------------------
static inline int cdiv(long a, long b) { return (int)((a + b - 1) / b); }

extern "C" void kernel_launch(void* const* d_in, const int* in_sizes, int n_in,
                              void* d_out, int out_size, void* d_ws, size_t ws_size,
                              hipStream_t stream) {
  const float* x   = (const float*)d_in[0];
  const int*   ei  = (const int*)d_in[1];
  const float* ea  = (const float*)d_in[2];
  const int*   e2e = (const int*)d_in[3];
  const int*   n2n = (const int*)d_in[4];
  const float* W1  = (const float*)d_in[5];
  const float* b1  = (const float*)d_in[6];
  const float* W2  = (const float*)d_in[7];
  const float* b2  = (const float*)d_in[8];
  const float* Wq  = (const float*)d_in[9];
  const float* Wk  = (const float*)d_in[10];
  const float* Wv  = (const float*)d_in[11];
  const float* Wq2 = (const float*)d_in[12];
  const float* Wk2 = (const float*)d_in[13];
  const float* Wv2 = (const float*)d_in[14];

  const int N  = in_sizes[0];
  const int E  = in_sizes[1] / 2;
  const int EE = in_sizes[3] / 2;
  const int C  = in_sizes[4] / 2;

  char* ws = (char*)d_ws;
  size_t off = 0;
  auto take = [&](size_t bytes) -> size_t {
    off = (off + 255) & ~(size_t)255;
    size_t o = off;
    off += bytes;
    return o;
  };

  __bf16* pW2 = (__bf16*)(ws + take((size_t)8  * 512 * 2));
  __bf16* pWq = (__bf16*)(ws + take((size_t)32 * 512 * 2));
  __bf16* pWk = (__bf16*)(ws + take((size_t)32 * 512 * 2));
  __bf16* pWv = (__bf16*)(ws + take((size_t)32 * 512 * 2));
  float*  coef = (float*)(ws + take(8 * sizeof(float)));
  float*  qb  = (float*)(ws + take((size_t)E * HID * 4));
  float*  kb  = (float*)(ws + take((size_t)E * HID * 4));
  float*  vb  = (float*)(ws + take((size_t)E * HID * 4));
  float*  logits = (float*)(ws + take((size_t)EE * 4 * 4));
  size_t zero_begin = (off + 255) & ~(size_t)255;
  unsigned* segmax = (unsigned*)(ws + take((size_t)E * 4 * 4));
  float*    segsum = (float*)(ws + take((size_t)E * 4 * 4));
  unsigned* nmax   = (unsigned*)(ws + take((size_t)N * 4 * 4));
  float*    nsum   = (float*)(ws + take((size_t)N * 4 * 4));
  float*    agg    = (float*)(ws + take((size_t)E * HID * 4));
  size_t zero_end = off;

  float* out = (float*)d_out;

  // zero accumulators (key 0 == -NaN sentinel for max buffers, 0.0f for sums)
  hipMemsetAsync(ws + zero_begin, 0, zero_end - zero_begin, stream);
  hipMemsetAsync(out, 0, (size_t)N * sizeof(float), stream);

  // pack weights into WMMA B-fragment layout
  pack_b_kernel<<<cdiv(8 * 512, 256),  256, 0, stream>>>(W2, pW2, 32);
  pack_b_kernel<<<cdiv(32 * 512, 256), 256, 0, stream>>>(Wq, pWq, 128);
  pack_b_kernel<<<cdiv(32 * 512, 256), 256, 0, stream>>>(Wk, pWk, 128);
  pack_b_kernel<<<cdiv(32 * 512, 256), 256, 0, stream>>>(Wv, pWv, 128);
  coef_kernel<<<1, 32, 0, stream>>>(Wq2, Wk2, Wv2, coef);

  // fused MLP + Q/K/V projections (WMMA)
  fused_mlp_qkv<<<cdiv(E, 128), 256, 0, stream>>>(x, ei, ea, W1, b1, pW2, b2,
                                                  pWq, pWk, pWv, qb, kb, vb, E);

  // edge-to-edge attention (segment softmax + aggregate)
  attn_logits<<<cdiv(EE, 8), 256, 0, stream>>>(qb, kb, e2e, logits, segmax, EE);
  attn_exp<<<cdiv((long)EE * 4, 256), 256, 0, stream>>>(e2e, logits, segmax, segsum, EE);
  attn_agg<<<cdiv(EE, 8), 256, 0, stream>>>(vb, e2e, logits, segsum, agg, EE);
  uef_mean<<<cdiv(E, 256), 256, 0, stream>>>(agg, out + N, E);

  // node-to-node attention (scalar-collapsed)
  n2n_max<<<cdiv(C, 256), 256, 0, stream>>>(x, n2n, coef, nmax, C);
  n2n_sum<<<cdiv(C, 256), 256, 0, stream>>>(x, n2n, coef, nmax, nsum, C);
  n2n_msg<<<cdiv(C, 256), 256, 0, stream>>>(x, n2n, coef, nmax, nsum, out, C);
}
